// SPIN_87505663688950
// MI455X (gfx1250) — matrix operand order repaired and checked by reading, not compiled, not run on hardware
//
#include <hip/hip_runtime.h>

typedef _Float16 h16;
typedef __attribute__((ext_vector_type(16))) _Float16 v16h;
typedef __attribute__((ext_vector_type(8)))  float    v8f;

#define BB   2
#define CC   64
#define HH   256
#define WW   256
#define SSZ  16
#define NHT  16
#define NWT  16
#define NST  256      // NHT*NWT supertokens
#define NPIX 65536    // HH*WW

#define APIX_STRIDE 72     // halfs; 144B rows: 16B aligned, conflict-free
#define PFC_STRIDE  264    // halfs; 528B rows: 16B aligned, conflict-free
#define WT_OFF      (256 * APIX_STRIDE * 2)           // 36864
#define DENL_OFF    (WT_OFF + 16 * PFC_STRIDE * 2)    // 45312
#define REDC_OFF    (DENL_OFF + 64)                   // 45376
#define SMEM0       (REDC_OFF + 4 * 32 * 8 * 4)       // 49472 (MODE 0)
#define SMEM1       WT_OFF                            // 36864 (MODE 1)
#define TILE_HALFS  16384                             // 256 px * 64 ch

// ---------------- gfx1250 async global->LDS copy (16B per lane) -----------
__device__ __forceinline__ void SPIN_async_b128(unsigned lds_off, const void* g) {
  asm volatile("global_load_async_to_lds_b128 %0, %1, off"
               :: "v"(lds_off), "v"(g) : "memory");
}
__device__ __forceinline__ void SPIN_wait_async() {
  asm volatile("s_wait_asynccnt 0x0" ::: "memory");
}

// ---------------------------------------------------------------- zero fill
__global__ void SPIN_zero_f4(float4* __restrict__ p, int n4) {
  int i = blockIdx.x * blockDim.x + threadIdx.x;
  if (i < n4) p[i] = make_float4(0.f, 0.f, 0.f, 0.f);
}

// ------------------------------------------------- per-supertoken block mean
// grid = B*NST, block = 256 (1 thread = 1 pixel); optionally emits the two
// f16 tile layouts consumed by the async staging path.
__global__ void SPIN_prep(const float* __restrict__ x, float* __restrict__ cent,
                          h16* __restrict__ pfT, h16* __restrict__ pfTc, int writePF) {
  int blk = blockIdx.x;
  int b = blk / NST, s = blk % NST;
  int sy = s / NWT, sx = s % NWT;
  int p = threadIdx.x;
  int ly = p >> 4, lx = p & 15;
  const float* xb = x + ((size_t)b * CC) * NPIX + (size_t)(sy * SSZ + ly) * WW + (sx * SSZ + lx);

  __shared__ float partial[8][CC];
  int wid = threadIdx.x >> 5, lane = threadIdx.x & 31;
  h16 hv[CC] __attribute__((aligned(16)));
  #pragma unroll
  for (int c = 0; c < CC; ++c) {
    float t = xb[(size_t)c * NPIX];
    hv[c] = (h16)t;
    if (writePF) pfTc[(size_t)blk * TILE_HALFS + c * 256 + p] = hv[c];   // [c][p]
    float v = t;
    v += __shfl_xor(v, 1);  v += __shfl_xor(v, 2);  v += __shfl_xor(v, 4);
    v += __shfl_xor(v, 8);  v += __shfl_xor(v, 16);
    if (lane == (c & 31)) partial[wid][c] = v;   // all lanes hold wave sum
  }
  if (writePF) {                                  // [p][c], 128B/thread
    uint4* dst = (uint4*)(pfT + (size_t)blk * TILE_HALFS + p * CC);
    #pragma unroll
    for (int q = 0; q < 8; ++q) dst[q] = ((uint4*)hv)[q];
  }
  __syncthreads();
  if (p < CC) {
    float ssum = 0.f;
    #pragma unroll
    for (int w = 0; w < 8; ++w) ssum += partial[w][p];
    cent[(size_t)blk * CC + p] = ssum * (1.0f / 256.0f);
  }
}

// ------------------------------------ centroid finalize: f16 copy + ||c||^2
__global__ void SPIN_centfin(const float* __restrict__ cent,
                             const float* __restrict__ num,
                             const float* __restrict__ den,
                             h16* __restrict__ centh, float* __restrict__ cnorm,
                             int useAcc) {
  int blk = blockIdx.x, t = threadIdx.x;
  int idx = blk * CC + t;
  float c = useAcc ? (num[idx] / (den[blk] + 1e-16f)) : cent[idx];
  centh[idx] = (h16)c;
  __shared__ float red[CC];
  red[t] = c * c;
  __syncthreads();
  for (int off = 32; off > 0; off >>= 1) {
    if (t < off) red[t] += red[t + off];
    __syncthreads();
  }
  if (t == 0) cnorm[blk] = red[0];
}

// ---------------------------------------------------------------- iteration
// MODE 0: accumulate num/den (centroid update GEMM); MODE 1: write affinity.
// ASYNC 1: tiles staged via global_load_async_to_lds_b128 from pfT/pfTc.
template <int MODE, int ASYNC>
__global__ void SPIN_iter(const float* __restrict__ x,
                          const h16* __restrict__ centh,
                          const float* __restrict__ cnorm,
                          float* __restrict__ num, float* __restrict__ den,
                          float* __restrict__ out,
                          const h16* __restrict__ pfT, const h16* __restrict__ pfTc) {
  extern __shared__ char smem[];
  h16*   Apix = (h16*)smem;                    // [256][72]  phase 1
  h16*   pfC  = (h16*)smem;                    // [64][264]  phase 2 (overlay)
  h16*   wT   = (h16*)(smem + WT_OFF);         // [16][264]
  float* denL = (float*)(smem + DENL_OFF);     // [16]
  float* redC = (float*)(smem + REDC_OFF);     // [4][32][8]

  int blk = blockIdx.x;
  int b = blk / NST, s = blk % NST;
  int sy = s / NWT, sx = s % NWT;
  int tid  = threadIdx.x;
  int lane = tid & 31, wid = tid >> 5;
  int p = tid, ly = p >> 4, lx = p & 15;
  const float* xb = x + ((size_t)b * CC) * NPIX + (size_t)(sy * SSZ + ly) * WW + (sx * SSZ + lx);

  // 3x3 candidate supertokens (uniform per block)
  int candS[9]; float cnS[9]; bool vK[9];
  #pragma unroll
  for (int k = 0; k < 9; ++k) {
    int cy = sy + (k / 3) - 1, cx = sx + (k % 3) - 1;
    bool v = (cy >= 0) && (cy < NHT) && (cx >= 0) && (cx < NWT);
    vK[k] = v;
    candS[k] = v ? (cy * NWT + cx) : s;        // invalid -> center (masked)
    cnS[k] = cnorm[b * NST + candS[k]];
  }

  if (MODE == 0 && tid < 16) denL[tid] = 0.f;

  // ---- phase 1: stage tile as [pixel][channel] f16
  if (ASYNC) {
    const char* g = (const char*)(pfT + (size_t)blk * TILE_HALFS);
    unsigned abase = (unsigned)(uintptr_t)Apix;
    #pragma unroll
    for (int j = 0; j < 8; ++j) {
      int chunk = j * 256 + tid;                       // 16B chunks, coalesced
      SPIN_async_b128(abase + (chunk >> 3) * 144 + (chunk & 7) * 16,
                      g + (size_t)chunk * 16);
    }
    SPIN_wait_async();
  } else {
    #pragma unroll 8
    for (int c = 0; c < CC; ++c) Apix[p * APIX_STRIDE + c] = (h16)xb[(size_t)c * NPIX];
  }
  __syncthreads();

  // A fragments: candidate centroids (M = cands, K = channels)
  int m    = lane & 15;
  int offA = (lane < 16) ? 0 : 16;   // A lane-half K split (16/16)
  int offB = (lane < 16) ? 0 : 16;   // B lane-half K split (16/16)
  const h16* cb = centh + (size_t)(b * NST + candS[m < 9 ? m : 4]) * CC;
  v16h afr[2];
  #pragma unroll
  for (int kc = 0; kc < 2; ++kc) {
    int kb = kc * 32 + offA;
    #pragma unroll
    for (int j = 0; j < 8; ++j) { afr[kc][j] = cb[kb + j]; afr[kc][8 + j] = cb[kb + 8 + j]; }
  }

  // logits via WMMA + 9-way softmax; each wave owns 2 groups of 16 pixels
  float wv[2][9];
  #pragma unroll
  for (int g2 = 0; g2 < 2; ++g2) {
    int g  = wid * 2 + g2;
    int pp = g * 16 + (lane & 15);
    v8f acc = {0.f, 0.f, 0.f, 0.f, 0.f, 0.f, 0.f, 0.f};
    #pragma unroll
    for (int kc = 0; kc < 2; ++kc) {
      v16h bfr;
      int kb = kc * 32 + offB;
      #pragma unroll
      for (int j = 0; j < 16; ++j) bfr[j] = Apix[pp * APIX_STRIDE + kb + j];
      acc = __builtin_amdgcn_wmma_f32_16x16x32_f16(false, afr[kc], false, bfr,
                                                   (short)0, acc, false, false);
    }
    float lg[8];
    #pragma unroll
    for (int r = 0; r < 8; ++r) {
      int mk = r + ((lane < 16) ? 0 : 8);
      lg[r] = (mk < 9 && vK[mk]) ? (2.f * acc[r] - cnS[mk]) : -1e30f;
    }
    float l8 = __shfl_xor(lg[0], 16);        // partner lane holds cand 8
    float mx = l8;
    #pragma unroll
    for (int r = 0; r < 8; ++r) mx = fmaxf(mx, lg[r]);
    float e[9], ssum = 0.f;
    #pragma unroll
    for (int r = 0; r < 8; ++r) { e[r] = __expf(lg[r] - mx); ssum += e[r]; }
    e[8] = __expf(l8 - mx); ssum += e[8];
    float inv = 1.f / ssum;
    #pragma unroll
    for (int k = 0; k < 9; ++k) wv[g2][k] = (lane < 16) ? e[k] * inv : 0.f;
  }
  __syncthreads();   // everyone done reading Apix

  if (MODE == 1) {
    if (lane < 16) {
      #pragma unroll
      for (int g2 = 0; g2 < 2; ++g2) {
        int g   = wid * 2 + g2;
        int pix = (sy * SSZ + g) * WW + sx * SSZ + lane;
        #pragma unroll
        for (int k = 0; k < 9; ++k)
          if (vK[k]) out[((size_t)(b * NST) + candS[k]) * NPIX + pix] = wv[g2][k];
      }
    }
  } else {
    // den: butterfly-reduce each candidate weight over the block
    #pragma unroll
    for (int k = 0; k < 9; ++k) {
      float v = wv[0][k] + wv[1][k];                 // lanes>=16 contribute 0
      v += __shfl_xor(v, 1);  v += __shfl_xor(v, 2); v += __shfl_xor(v, 4);
      v += __shfl_xor(v, 8);  v += __shfl_xor(v, 16);
      if (lane == 0) atomicAdd(&denL[k], v);
    }
    // stage wT[cand][pixel] and pfC[ch][pixel] (K = pixels contiguous)
    if (lane < 16) {
      #pragma unroll
      for (int g2 = 0; g2 < 2; ++g2) {
        int pp = (wid * 2 + g2) * 16 + lane;
        #pragma unroll
        for (int k = 0; k < 9; ++k) wT[k * PFC_STRIDE + pp] = (h16)wv[g2][k];
        #pragma unroll
        for (int k = 9; k < 16; ++k) wT[k * PFC_STRIDE + pp] = (h16)0.f;
      }
    }
    if (ASYNC) {
      const char* g = (const char*)(pfTc + (size_t)blk * TILE_HALFS);
      unsigned cbase = (unsigned)(uintptr_t)pfC;
      #pragma unroll
      for (int j = 0; j < 8; ++j) {
        int chunk = j * 256 + tid;                     // 32 chunks per channel row
        SPIN_async_b128(cbase + (chunk >> 5) * (PFC_STRIDE * 2) + (chunk & 31) * 16,
                        g + (size_t)chunk * 16);
      }
      SPIN_wait_async();
    } else {
      #pragma unroll 8
      for (int c = 0; c < CC; ++c) pfC[c * PFC_STRIDE + p] = (h16)xb[(size_t)c * NPIX];
    }
    __syncthreads();

    // num GEMM: [16 cand x 256 px] x [256 px x 64 ch], WMMA over 8 k-chunks
    int nt  = wid & 3;
    int kc0 = (wid < 4) ? 0 : 4;
    int ch  = nt * 16 + (lane & 15);
    v8f acc = {0.f, 0.f, 0.f, 0.f, 0.f, 0.f, 0.f, 0.f};
    #pragma unroll
    for (int kci = 0; kci < 4; ++kci) {
      int kb = (kc0 + kci) * 32;
      v16h afr2, bfr2;
      #pragma unroll
      for (int j = 0; j < 8; ++j) {
        afr2[j]     = wT[m * PFC_STRIDE + kb + offA + j];
        afr2[8 + j] = wT[m * PFC_STRIDE + kb + offA + 8 + j];
      }
      #pragma unroll
      for (int j = 0; j < 16; ++j) bfr2[j] = pfC[ch * PFC_STRIDE + kb + offB + j];
      acc = __builtin_amdgcn_wmma_f32_16x16x32_f16(false, afr2, false, bfr2,
                                                   (short)0, acc, false, false);
    }
    if (wid >= 4) {
      #pragma unroll
      for (int r = 0; r < 8; ++r) redC[((wid & 3) * 32 + lane) * 8 + r] = acc[r];
    }
    __syncthreads();
    if (wid < 4) {
      #pragma unroll
      for (int r = 0; r < 8; ++r) {
        float v = acc[r] + redC[(nt * 32 + lane) * 8 + r];
        int mk = r + ((lane < 16) ? 0 : 8);
        if (mk < 9 && vK[mk])
          atomicAdd(&num[((size_t)(b * NST) + candS[mk]) * CC + ch], v);
      }
    }
    __syncthreads();
    if (tid < 9 && vK[tid]) atomicAdd(&den[b * NST + candS[tid]], denL[tid]);
  }
}

// ------------------------------------------------------------------ launch
extern "C" void kernel_launch(void* const* d_in, const int* in_sizes, int n_in,
                              void* d_out, int out_size, void* d_ws, size_t ws_size,
                              hipStream_t stream) {
  (void)in_sizes; (void)n_in; (void)out_size;
  const float* x = (const float*)d_in[0];
  float* out = (float*)d_out;

  // workspace carve
  float* cent  = (float*)d_ws;                 // 32768 f32
  float* num   = cent + BB * NST * CC;         // 32768 f32
  float* den   = num  + BB * NST * CC;         // 512 f32
  float* cnorm = den  + BB * NST;              // 512 f32
  h16*   centh = (h16*)(cnorm + BB * NST);     // 32768 f16
  const size_t baseBytes = (size_t)(BB * NST * CC) * 8 + (size_t)(BB * NST) * 8
                         + (size_t)(BB * NST * CC) * 2;          // 331776
  const size_t pfHalfs = (size_t)BB * NST * TILE_HALFS;          // 8M halfs
  h16* pfT  = (h16*)((char*)d_ws + baseBytes);                   // [blk][p][c]
  h16* pfTc = pfT + pfHalfs;                                     // [blk][c][p]
  const bool useAsync = ws_size >= baseBytes + 2 * pfHalfs * sizeof(h16);

  const int nBlocks = BB * NST;                // 512 tiles

  { // zero num+den (contiguous)
    int n4 = (BB * NST * CC + BB * NST) / 4;
    SPIN_zero_f4<<<(n4 + 255) / 256, 256, 0, stream>>>((float4*)num, n4);
  }
  SPIN_prep<<<nBlocks, 256, 0, stream>>>(x, cent, pfT, pfTc, useAsync ? 1 : 0);
  SPIN_centfin<<<nBlocks, CC, 0, stream>>>(cent, num, den, centh, cnorm, 0);
  if (useAsync)
    SPIN_iter<0, 1><<<nBlocks, 256, SMEM0, stream>>>(x, centh, cnorm, num, den, nullptr, pfT, pfTc);
  else
    SPIN_iter<0, 0><<<nBlocks, 256, SMEM0, stream>>>(x, centh, cnorm, num, den, nullptr, pfT, pfTc);
  SPIN_centfin<<<nBlocks, CC, 0, stream>>>(cent, num, den, centh, cnorm, 1);
  { // zero the 128 MiB output
    int n4 = (BB * NST * NPIX) / 4;
    SPIN_zero_f4<<<(n4 + 255) / 256, 256, 0, stream>>>((float4*)out, n4);
  }
  if (useAsync)
    SPIN_iter<1, 1><<<nBlocks, 256, SMEM1, stream>>>(x, centh, cnorm, nullptr, nullptr, out, pfT, pfTc);
  else
    SPIN_iter<1, 0><<<nBlocks, 256, SMEM1, stream>>>(x, centh, cnorm, nullptr, nullptr, out, pfT, pfTc);
}
